// ScannedRNN_85031762526209
// MI455X (gfx1250) — matrix operand order repaired and checked by reading, not compiled, not run on hardware
//
#include <hip/hip_runtime.h>

#define T_ 512
#define B_ 128
#define H_ 512
#define NWG 32u            // 32 workgroups x 16 cols = 512 h columns
#define PACKED_ELEMS (2u*3u*32u*16u*32u*16u)   // 1,572,864 bf16 = 3MB

typedef __attribute__((ext_vector_type(16))) __bf16 v16bf;
typedef __attribute__((ext_vector_type(8)))  float  v8f;

// ---------------------------------------------------------------------------
// Kernel 1: pack Wi/Wh (f32 [H,3H] row-major) into bf16 WMMA B-fragment order.
// Flat layout: idx = ((((w*3+g)*32 + tn)*16 + c)*32 + lane)*16 + i
//   col = 512*g + 16*tn + (lane&15) ; K = 32*c + 8*(lane>>4) + (i&7) + 16*(i>>3)
// Matches CDNA5 16-bit B-matrix layout (lanes 0-15: K 0..7,16..23, lanes 16-31: K 8..15,24..31).
// Also zeroes the grid-barrier counter (ws is not re-initialized between replays).
// ---------------------------------------------------------------------------
__global__ __launch_bounds__(256)
void pack_weights(const float* __restrict__ Wi, const float* __restrict__ Wh,
                  __bf16* __restrict__ packed, unsigned* __restrict__ cnt) {
  unsigned idx = blockIdx.x * 256u + threadIdx.x;
  if (idx == 0u) *cnt = 0u;
  if (idx >= PACKED_ELEMS) return;
  unsigned i  = idx & 15u;
  unsigned t1 = idx >> 4;
  unsigned l  = t1 & 31u;
  unsigned t2 = t1 >> 5;
  unsigned c  = t2 & 15u;
  unsigned t3 = t2 >> 4;
  unsigned tn = t3 & 31u;
  unsigned t4 = t3 >> 5;            // w*3+g in 0..5
  unsigned g  = t4 % 3u;
  unsigned w  = t4 / 3u;
  unsigned col = g * 512u + tn * 16u + (l & 15u);
  unsigned k   = c * 32u + (l >> 4) * 8u + (i & 7u) + (i >> 3) * 16u;
  const float* src = w ? Wh : Wi;
  packed[idx] = (__bf16)src[(size_t)k * 1536u + col];
}

// ---------------------------------------------------------------------------
// Kernel 2 (big-ws path): chip-wide parallel GEMM  Gi = X @ Wi + bi.
// Grid: (T*B/128) row-blocks x 32 col-tiles. Each wave: 16x16 tile x 3 gates.
// ---------------------------------------------------------------------------
__global__ __launch_bounds__(256)
void gi_gemm(const float* __restrict__ x, const float* __restrict__ bi,
             const __bf16* __restrict__ packed, float* __restrict__ gi) {
  __shared__ __bf16 ldsW[3u * 16u * 32u * 16u];   // Wi blocks only: 48 KB

  const unsigned tn  = blockIdx.x & 31u;
  const unsigned rb  = blockIdx.x >> 5;           // 0..511 (128 rows each)
  const unsigned tid = threadIdx.x;
  const unsigned wave = tid >> 5, lane = tid & 31u;
  const unsigned m  = wave * 16u;
  const unsigned j  = tn * 16u;
  const unsigned hb = lane >> 4;
  const unsigned ln = lane & 15u;

  {
    const uint4* src = (const uint4*)packed;
    uint4*       dst = (uint4*)ldsW;
    for (unsigned b = 0; b < 3u; ++b) {
      unsigned goff = (b * 32u + tn) * 1024u;
      unsigned loff = b * 1024u;
      for (unsigned q = tid; q < 1024u; q += 256u)
        dst[loff + q] = src[goff + q];
    }
  }
  __syncthreads();

  const float bir = bi[j + ln];
  const float biz = bi[512u + j + ln];
  const float bin = bi[1024u + j + ln];

  const unsigned row_a = rb * 128u + m + ln;
  const float* xrow = x + (size_t)row_a * H_;

  v8f aR = {}, aZ = {}, aN = {};
#pragma unroll 4
  for (unsigned c = 0; c < 16u; ++c) {
    const unsigned kb = c * 32u + hb * 8u;
    v16bf xa;
    {
      const float4 f0 = *(const float4*)(xrow + kb);
      const float4 f1 = *(const float4*)(xrow + kb + 4);
      const float4 f2 = *(const float4*)(xrow + kb + 16);
      const float4 f3 = *(const float4*)(xrow + kb + 20);
      xa[0]=(__bf16)f0.x; xa[1]=(__bf16)f0.y; xa[2]=(__bf16)f0.z; xa[3]=(__bf16)f0.w;
      xa[4]=(__bf16)f1.x; xa[5]=(__bf16)f1.y; xa[6]=(__bf16)f1.z; xa[7]=(__bf16)f1.w;
      xa[8]=(__bf16)f2.x; xa[9]=(__bf16)f2.y; xa[10]=(__bf16)f2.z; xa[11]=(__bf16)f2.w;
      xa[12]=(__bf16)f3.x; xa[13]=(__bf16)f3.y; xa[14]=(__bf16)f3.z; xa[15]=(__bf16)f3.w;
    }
    const v16bf bR = *(const v16bf*)&ldsW[(0u * 16u + c) * 512u + lane * 16u];
    const v16bf bZ = *(const v16bf*)&ldsW[(1u * 16u + c) * 512u + lane * 16u];
    const v16bf bN = *(const v16bf*)&ldsW[(2u * 16u + c) * 512u + lane * 16u];
    aR = __builtin_amdgcn_wmma_f32_16x16x32_bf16(false, xa, false, bR, (short)0, aR, false, false);
    aZ = __builtin_amdgcn_wmma_f32_16x16x32_bf16(false, xa, false, bZ, (short)0, aZ, false, false);
    aN = __builtin_amdgcn_wmma_f32_16x16x32_bf16(false, xa, false, bN, (short)0, aN, false, false);
  }

#pragma unroll
  for (int r = 0; r < 8; ++r) {
    const size_t row  = (size_t)(rb * 128u + m + hb * 8u + r);
    const size_t base = row * 1536u + j + ln;
    gi[base]          = aR[r] + bir;
    gi[base + 512u]   = aZ[r] + biz;
    gi[base + 1024u]  = aN[r] + bin;
  }
}

// ---------------------------------------------------------------------------
// Kernel 3 (big-ws path): persistent GRU scan with precomputed Gi.
// Per step/wave: 48 WMMAs (gh only); Gi tile preloaded into the WMMA C
// accumulators *before* the grid barrier to hide load latency.
// ---------------------------------------------------------------------------
__global__ __launch_bounds__(256)
void gru_persistent_pregi(const unsigned char* __restrict__ resets,
                          const float* __restrict__ h0, const float* __restrict__ bhn,
                          const __bf16* __restrict__ packed, const float* __restrict__ gi,
                          __bf16* __restrict__ hbuf, unsigned* __restrict__ cnt,
                          float* __restrict__ out) {
  __shared__ __bf16 ldsW[3u * 16u * 32u * 16u];   // Wh blocks only: 48 KB

  const unsigned tn   = blockIdx.x;
  const unsigned tid  = threadIdx.x;
  const unsigned wave = tid >> 5, lane = tid & 31u;
  const unsigned m  = wave * 16u;
  const unsigned j  = tn * 16u;
  const unsigned hb = lane >> 4;
  const unsigned ln = lane & 15u;

  {
    const uint4* src = (const uint4*)packed;
    uint4*       dst = (uint4*)ldsW;
    for (unsigned b = 3; b < 6u; ++b) {
      unsigned goff = (b * 32u + tn) * 1024u;
      unsigned loff = (b - 3u) * 1024u;
      for (unsigned q = tid; q < 1024u; q += 256u)
        dst[loff + q] = src[goff + q];
    }
  }
  __syncthreads();

  const float bn = bhn[j + ln];

  v8f h;
#pragma unroll
  for (int r = 0; r < 8; ++r)
    h[r] = h0[(size_t)(m + hb * 8u + r) * H_ + j + ln];

  const unsigned row_a = m + ln;

  for (unsigned t = 0; t < T_; ++t) {
    // reset gating
    unsigned long long fl =
        *(const unsigned long long*)(resets + (size_t)t * B_ + m + hb * 8u);
#pragma unroll
    for (int r = 0; r < 8; ++r)
      if ((fl >> (8 * r)) & 0xffull) h[r] = 0.0f;

    // publish post-reset h
    __bf16* hout = hbuf + (size_t)(t & 1u) * (B_ * H_);
#pragma unroll
    for (int r = 0; r < 8; ++r)
      hout[(size_t)(m + hb * 8u + r) * H_ + j + ln] = (__bf16)h[r];

    // preload Gi tile (overlaps barrier latency); r/z go straight into C accums
    v8f aR, aZ, giN;
    {
      const float* girow = gi + (size_t)t * B_ * 1536u;
#pragma unroll
      for (int r = 0; r < 8; ++r) {
        const size_t base = (size_t)(m + hb * 8u + r) * 1536u + j + ln;
        aR[r]  = girow[base];
        aZ[r]  = girow[base + 512u];
        giN[r] = girow[base + 1024u];
      }
    }

    // grid barrier
    __threadfence();
    __syncthreads();
    if (tid == 0u) {
      atomicAdd(cnt, 1u);
      while (__hip_atomic_load(cnt, __ATOMIC_RELAXED, __HIP_MEMORY_SCOPE_AGENT) <
             NWG * (t + 1u))
        __builtin_amdgcn_s_sleep(1);
    }
    __syncthreads();
    __threadfence();

    const __bf16* hrow = hbuf + (size_t)(t & 1u) * (B_ * H_) + (size_t)row_a * H_;

    v8f aN = {};
#pragma unroll 4
    for (unsigned c = 0; c < 16u; ++c) {
      const unsigned kb = c * 32u + hb * 8u;
      v16bf ha;
      *((uint4*)&ha)     = *(const uint4*)(hrow + kb);
      *((uint4*)&ha + 1) = *(const uint4*)(hrow + kb + 16);
      const v16bf bR = *(const v16bf*)&ldsW[(0u * 16u + c) * 512u + lane * 16u];
      const v16bf bZ = *(const v16bf*)&ldsW[(1u * 16u + c) * 512u + lane * 16u];
      const v16bf bN = *(const v16bf*)&ldsW[(2u * 16u + c) * 512u + lane * 16u];
      aR = __builtin_amdgcn_wmma_f32_16x16x32_bf16(false, ha, false, bR, (short)0, aR, false, false);
      aZ = __builtin_amdgcn_wmma_f32_16x16x32_bf16(false, ha, false, bZ, (short)0, aZ, false, false);
      aN = __builtin_amdgcn_wmma_f32_16x16x32_bf16(false, ha, false, bN, (short)0, aN, false, false);
    }

    float* obase = out + ((size_t)t * B_) * H_;
#pragma unroll
    for (int r = 0; r < 8; ++r) {
      const float gr = 1.0f / (1.0f + __expf(-aR[r]));
      const float gz = 1.0f / (1.0f + __expf(-aZ[r]));
      const float na = giN[r] + gr * (aN[r] + bn);
      const float e  = __expf(-2.0f * na);
      const float gn = (1.0f - e) / (1.0f + e);          // tanh(na)
      const float nh = (1.0f - gz) * gn + gz * h[r];
      h[r] = nh;
      obase[(size_t)(m + hb * 8u + r) * H_ + j + ln] = nh;
    }
  }
}

// ---------------------------------------------------------------------------
// Kernel 4 (fallback, small ws): fused persistent GRU (x GEMM on critical path).
// ---------------------------------------------------------------------------
__global__ __launch_bounds__(256)
void gru_persistent(const float* __restrict__ x, const unsigned char* __restrict__ resets,
                    const float* __restrict__ h0, const float* __restrict__ bi,
                    const float* __restrict__ bhn, const __bf16* __restrict__ packed,
                    __bf16* __restrict__ hbuf, unsigned* __restrict__ cnt,
                    float* __restrict__ out) {
  __shared__ __bf16 ldsW[6u * 16u * 32u * 16u];   // 96 KB

  const unsigned tn   = blockIdx.x;
  const unsigned tid  = threadIdx.x;
  const unsigned wave = tid >> 5, lane = tid & 31u;
  const unsigned m  = wave * 16u;
  const unsigned j  = tn * 16u;
  const unsigned hb = lane >> 4;
  const unsigned ln = lane & 15u;

  {
    const uint4* src = (const uint4*)packed;
    uint4*       dst = (uint4*)ldsW;
    for (unsigned b = 0; b < 6u; ++b) {
      unsigned goff = (b * 32u + tn) * 1024u;
      unsigned loff = b * 1024u;
      for (unsigned q = tid; q < 1024u; q += 256u)
        dst[loff + q] = src[goff + q];
    }
  }
  __syncthreads();

  const float bir = bi[j + ln];
  const float biz = bi[512u + j + ln];
  const float bin = bi[1024u + j + ln];
  const float bn  = bhn[j + ln];

  v8f h;
#pragma unroll
  for (int r = 0; r < 8; ++r)
    h[r] = h0[(size_t)(m + hb * 8u + r) * H_ + j + ln];

  const unsigned row_a = m + ln;

  for (unsigned t = 0; t < T_; ++t) {
    unsigned long long fl =
        *(const unsigned long long*)(resets + (size_t)t * B_ + m + hb * 8u);
#pragma unroll
    for (int r = 0; r < 8; ++r)
      if ((fl >> (8 * r)) & 0xffull) h[r] = 0.0f;

    __bf16* hout = hbuf + (size_t)(t & 1u) * (B_ * H_);
#pragma unroll
    for (int r = 0; r < 8; ++r)
      hout[(size_t)(m + hb * 8u + r) * H_ + j + ln] = (__bf16)h[r];

    __threadfence();
    __syncthreads();
    if (tid == 0u) {
      atomicAdd(cnt, 1u);
      while (__hip_atomic_load(cnt, __ATOMIC_RELAXED, __HIP_MEMORY_SCOPE_AGENT) <
             NWG * (t + 1u))
        __builtin_amdgcn_s_sleep(1);
    }
    __syncthreads();
    __threadfence();

    const float*  xrow = x + ((size_t)t * B_ + row_a) * H_;
    const __bf16* hrow = hbuf + (size_t)(t & 1u) * (B_ * H_) + (size_t)row_a * H_;

    if (t + 1u < T_) {
      const char* nx = (const char*)(xrow + (size_t)B_ * H_);
      __builtin_prefetch(nx, 0, 1);
      __builtin_prefetch(nx + 512, 0, 1);
      __builtin_prefetch(nx + 1024, 0, 1);
      __builtin_prefetch(nx + 1536, 0, 1);
    }

    v8f air = {}, aiz = {}, ain = {}, ahr = {}, ahz = {}, ahn = {};

#pragma unroll 4
    for (unsigned c = 0; c < 16u; ++c) {
      const unsigned kb = c * 32u + hb * 8u;
      v16bf xa;
      {
        const float4 f0 = *(const float4*)(xrow + kb);
        const float4 f1 = *(const float4*)(xrow + kb + 4);
        const float4 f2 = *(const float4*)(xrow + kb + 16);
        const float4 f3 = *(const float4*)(xrow + kb + 20);
        xa[0]=(__bf16)f0.x; xa[1]=(__bf16)f0.y; xa[2]=(__bf16)f0.z; xa[3]=(__bf16)f0.w;
        xa[4]=(__bf16)f1.x; xa[5]=(__bf16)f1.y; xa[6]=(__bf16)f1.z; xa[7]=(__bf16)f1.w;
        xa[8]=(__bf16)f2.x; xa[9]=(__bf16)f2.y; xa[10]=(__bf16)f2.z; xa[11]=(__bf16)f2.w;
        xa[12]=(__bf16)f3.x; xa[13]=(__bf16)f3.y; xa[14]=(__bf16)f3.z; xa[15]=(__bf16)f3.w;
      }
      v16bf ha;
      *((uint4*)&ha)     = *(const uint4*)(hrow + kb);
      *((uint4*)&ha + 1) = *(const uint4*)(hrow + kb + 16);

      const v16bf bir_ = *(const v16bf*)&ldsW[(0u * 16u + c) * 512u + lane * 16u];
      const v16bf biz_ = *(const v16bf*)&ldsW[(1u * 16u + c) * 512u + lane * 16u];
      const v16bf bin_ = *(const v16bf*)&ldsW[(2u * 16u + c) * 512u + lane * 16u];
      const v16bf bhr_ = *(const v16bf*)&ldsW[(3u * 16u + c) * 512u + lane * 16u];
      const v16bf bhz_ = *(const v16bf*)&ldsW[(4u * 16u + c) * 512u + lane * 16u];
      const v16bf bhn_ = *(const v16bf*)&ldsW[(5u * 16u + c) * 512u + lane * 16u];

      air = __builtin_amdgcn_wmma_f32_16x16x32_bf16(false, xa, false, bir_, (short)0, air, false, false);
      aiz = __builtin_amdgcn_wmma_f32_16x16x32_bf16(false, xa, false, biz_, (short)0, aiz, false, false);
      ain = __builtin_amdgcn_wmma_f32_16x16x32_bf16(false, xa, false, bin_, (short)0, ain, false, false);
      ahr = __builtin_amdgcn_wmma_f32_16x16x32_bf16(false, ha, false, bhr_, (short)0, ahr, false, false);
      ahz = __builtin_amdgcn_wmma_f32_16x16x32_bf16(false, ha, false, bhz_, (short)0, ahz, false, false);
      ahn = __builtin_amdgcn_wmma_f32_16x16x32_bf16(false, ha, false, bhn_, (short)0, ahn, false, false);
    }

    float* obase = out + ((size_t)t * B_) * H_;
#pragma unroll
    for (int r = 0; r < 8; ++r) {
      const float gr = 1.0f / (1.0f + __expf(-(air[r] + bir + ahr[r])));
      const float gz = 1.0f / (1.0f + __expf(-(aiz[r] + biz + ahz[r])));
      const float na = ain[r] + bin + gr * (ahn[r] + bn);
      const float e  = __expf(-2.0f * na);
      const float gn = (1.0f - e) / (1.0f + e);
      const float nh = (1.0f - gz) * gn + gz * h[r];
      h[r] = nh;
      obase[(size_t)(m + hb * 8u + r) * H_ + j + ln] = nh;
    }
  }
}

// ---------------------------------------------------------------------------
extern "C" void kernel_launch(void* const* d_in, const int* in_sizes, int n_in,
                              void* d_out, int out_size, void* d_ws, size_t ws_size,
                              hipStream_t stream) {
  (void)in_sizes; (void)n_in; (void)out_size;
  const float*         x      = (const float*)d_in[0];
  const unsigned char* resets = (const unsigned char*)d_in[1];
  const float*         h0     = (const float*)d_in[2];
  const float*         Wi     = (const float*)d_in[3];
  const float*         bi     = (const float*)d_in[4];
  const float*         Wh     = (const float*)d_in[5];
  const float*         bhn    = (const float*)d_in[6];
  float*               out    = (float*)d_out;

  unsigned char* ws = (unsigned char*)d_ws;
  unsigned* cnt     = (unsigned*)ws;                                     // 256 B slot
  __bf16*   hbuf    = (__bf16*)(ws + 256);                               // 256 KB
  __bf16*   packed  = (__bf16*)(ws + 256 + (size_t)2 * B_ * H_ * 2);     // 3 MB
  float*    gibuf   = (float*)(ws + 256 + (size_t)2 * B_ * H_ * 2 + (size_t)PACKED_ELEMS * 2);

  const size_t need_big = 256 + (size_t)2 * B_ * H_ * 2 + (size_t)PACKED_ELEMS * 2 +
                          (size_t)T_ * B_ * 1536 * sizeof(float);        // ~406 MB

  pack_weights<<<(PACKED_ELEMS + 255u) / 256u, 256, 0, stream>>>(Wi, Wh, packed, cnt);

  if (ws_size >= need_big) {
    // Chip-wide parallel input-gate GEMM, then lean sequential scan.
    gi_gemm<<<dim3((T_ * B_ / 128) * 32), 256, 0, stream>>>(x, bi, packed, gibuf);
    gru_persistent_pregi<<<NWG, 256, 0, stream>>>(resets, h0, bhn, packed, gibuf,
                                                  hbuf, cnt, out);
  } else {
    gru_persistent<<<NWG, 256, 0, stream>>>(x, resets, h0, bi, bhn, packed,
                                            hbuf, cnt, out);
  }
}
